// PolicyNetworkPad_19542101197119
// MI455X (gfx1250) — 16-bit and 8-bit matrix operand order repaired and checked by reading, compiled, not run on hardware
//
#include <hip/hip_runtime.h>

typedef __attribute__((ext_vector_type(16))) _Float16 v16h;
typedef __attribute__((ext_vector_type(8)))  _Float16 v8h;
typedef __attribute__((ext_vector_type(8)))  float    v8f;
typedef __attribute__((ext_vector_type(16))) int      v16i;

constexpr int BB = 16, TT = 30, NSELC = 5, NCLSC = 10;

__device__ __forceinline__ unsigned char f32_to_e4m3(float f) {
  unsigned sign = (__float_as_uint(f) >> 24) & 0x80u;
  float a = fminf(fabsf(f), 448.f);
  unsigned ur = __float_as_uint(a) + 0x00080000u;
  int E = (int)((ur >> 23) & 0xffu) - 120;
  unsigned mant = (ur >> 20) & 7u;
  unsigned norm = ((unsigned)E << 3) | mant;
  int subm = (int)(a * 512.f + 0.5f);
  unsigned res = (E <= 0) ? (unsigned)subm : norm;
  return (unsigned char)(sign | res);
}

__global__ void k_cvt_weights(const float* __restrict__ w1,
                              const float* __restrict__ w2,
                              _Float16* __restrict__ w1h,
                              unsigned char* __restrict__ w2q) {
  int tid = blockIdx.x * blockDim.x + threadIdx.x;
  if (tid < 64 * 32) {
    int oc = tid >> 5, k = tid & 31;
    w1h[tid] = (_Float16)(k < 27 ? w1[oc * 27 + k] : 0.f);
  }
  int t2 = tid - 64 * 32;
  if (t2 >= 0 && t2 < 128 * 640) {
    int n = t2 / 640, k = t2 % 640;
    int tap = k >> 6, c = k & 63;
    float v = 0.f;
    if (tap < 9) v = w2[((n * 64 + c) * 3 + (tap / 3)) * 3 + (tap % 3)];
    w2q[t2] = f32_to_e4m3(v);
  }
}

__global__ void k_skim_resize(const float* __restrict__ f64,
                              float* __restrict__ out) {
  int tid = blockIdx.x * blockDim.x + threadIdx.x;
  if (tid >= BB * 3 * 64 * 64) return;
  int w  = tid & 63;
  int oh = (tid >> 6) & 63;
  int ch = (tid >> 12) % 3;
  int b  = tid / (3 * 64 * 64);
  const float scale = 1.f / 30.f;
  float center = (oh + 0.5f) * 30.f - 0.5f;
  int lo = (int)ceilf(center - 30.f);  if (lo < 0) lo = 0;
  int hi = (int)floorf(center + 30.f); if (hi > TT * 64 - 1) hi = TT * 64 - 1;
  float acc = 0.f, wsum = 0.f;
  for (int i = lo; i <= hi; ++i) {
    float wt = 1.f - fabsf((float)i - center) * scale;
    if (wt <= 0.f) continue;
    int t = i >> 6, h = i & 63;
    acc  += wt * f64[((((size_t)b * TT + t) * 3 + ch) * 64 + h) * 64 + w];
    wsum += wt;
  }
  out[tid] = acc / wsum;
}

__global__ void k_sconv1(const float* __restrict__ x, const float* __restrict__ w,
                         float* __restrict__ y) {
  int tid = blockIdx.x * blockDim.x + threadIdx.x;
  if (tid >= BB * 32 * 32 * 32) return;
  int ox = tid & 31, oy = (tid >> 5) & 31, oc = (tid >> 10) & 31, b = tid >> 15;
  float acc = 0.f;
  for (int c = 0; c < 3; ++c)
    for (int ky = 0; ky < 3; ++ky) {
      int iy = 2 * oy + ky;
      if (iy >= 64) continue;
      for (int kx = 0; kx < 3; ++kx) {
        int ix = 2 * ox + kx;
        if (ix >= 64) continue;
        acc += x[(((size_t)b * 3 + c) * 64 + iy) * 64 + ix] *
               w[((oc * 3 + c) * 3 + ky) * 3 + kx];
      }
    }
  y[tid] = fmaxf(acc, 0.f);
}

__global__ void k_sconv2(const float* __restrict__ x, const float* __restrict__ w,
                         float* __restrict__ y) {
  int tid = blockIdx.x * blockDim.x + threadIdx.x;
  if (tid >= BB * 64 * 16 * 16) return;
  int ox = tid & 15, oy = (tid >> 4) & 15, oc = (tid >> 8) & 63, b = tid >> 14;
  float acc = 0.f;
  for (int c = 0; c < 32; ++c)
    for (int ky = 0; ky < 3; ++ky) {
      int iy = 2 * oy + ky;
      if (iy >= 32) continue;
      for (int kx = 0; kx < 3; ++kx) {
        int ix = 2 * ox + kx;
        if (ix >= 32) continue;
        acc += x[(((size_t)b * 32 + c) * 32 + iy) * 32 + ix] *
               w[((oc * 32 + c) * 3 + ky) * 3 + kx];
      }
    }
  y[tid] = fmaxf(acc, 0.f);
}

__global__ void k_policy(const float* __restrict__ h2, const float* __restrict__ pw,
                         const float* __restrict__ pb, int* __restrict__ sel_idx) {
  __shared__ float fea[64];
  int b = blockIdx.x, c = threadIdx.x;
  const float* p = h2 + ((size_t)b * 64 + c) * 256;
  float s = 0.f;
  for (int i = 0; i < 256; ++i) s += p[i];
  float f = s * (1.f / 256.f);
  fea[c] = f > 0.f ? f : 0.f;
  __syncthreads();
  if (c == 0) {
    float logits[TT];
    for (int t = 0; t < TT; ++t) {
      float a = pb[t];
      for (int cc = 0; cc < 64; ++cc) a += fea[cc] * pw[cc * TT + t];
      logits[t] = a;
    }
    int sel[NSELC];
    unsigned used = 0u;
    for (int s5 = 0; s5 < NSELC; ++s5) {
      int best = 0; float bv = -3.4e38f;
      for (int t = 0; t < TT; ++t)
        if (!((used >> t) & 1u) && logits[t] > bv) { bv = logits[t]; best = t; }
      used |= 1u << best;
      sel[s5] = best;
    }
    for (int i = 1; i < NSELC; ++i) {
      int v = sel[i], j = i - 1;
      while (j >= 0 && sel[j] > v) { sel[j + 1] = sel[j]; --j; }
      sel[j + 1] = v;
    }
    for (int s5 = 0; s5 < NSELC; ++s5) sel_idx[b * NSELC + s5] = sel[s5];
  }
}

__global__ void k_eval_resize(const float* __restrict__ f224,
                              const int* __restrict__ sel,
                              _Float16* __restrict__ out) {
  int tid = blockIdx.x * blockDim.x + threadIdx.x;
  if (tid >= BB * 3 * 224 * 224) return;
  int w  = tid % 224;
  int oh = (tid / 224) % 224;
  int ch = (tid / (224 * 224)) % 3;
  int b  = tid / (3 * 224 * 224);
  float center = (oh + 0.5f) * 5.f - 0.5f;
  int lo = (int)ceilf(center - 5.f);  if (lo < 0) lo = 0;
  int hi = (int)floorf(center + 5.f); if (hi > NSELC * 224 - 1) hi = NSELC * 224 - 1;
  float acc = 0.f, wsum = 0.f;
  for (int i = lo; i <= hi; ++i) {
    float wt = 1.f - fabsf((float)i - center) * 0.2f;
    if (wt <= 0.f) continue;
    int fr = i / 224, r = i % 224;
    int t  = sel[b * NSELC + fr];
    acc  += wt * f224[((((size_t)b * TT + t) * 3 + ch) * 224 + r) * 224 + w];
    wsum += wt;
  }
  out[tid] = (_Float16)(acc / wsum);
}

template <int KB>
__device__ __forceinline__ v16h conv1_stage(const _Float16* __restrict__ xb,
                                            int oy, int ox) {
  const bool oyok = oy < 111;
  const bool oxok = ox < 111;
  const size_t base = (size_t)(2 * oy) * 224 + 2 * ox;
  v16h av;
#pragma unroll
  for (int j = 0; j < 16; ++j) {
    int k = KB + j;
    _Float16 v = (_Float16)0.f;
    if (k < 27) {
      int c = k / 9, r = k % 9, ky = r / 3, kx = r % 3;
      bool ok = (ky < 2 || oyok) && (kx < 2 || oxok);
      if (ok) v = xb[base + (size_t)c * (224 * 224) + ky * 224 + kx];
    }
    av[j] = v;
  }
  return av;
}

__global__ void k_conv1_wmma(const _Float16* __restrict__ x,
                             const _Float16* __restrict__ w,
                             unsigned char* __restrict__ y) {
  constexpr int IH = 224, OH = 112, IC = 3, OC = 64;
  __shared__ __align__(32) _Float16 lA[16 * 32];

  const int lane = threadIdx.x;
  const int m0   = blockIdx.x * 16;

  const int mloc = lane >> 1;
  const int kloc = (lane & 1) * 16;
  {
    int m   = m0 + mloc;
    int b   = m / (OH * OH);
    int rem = m % (OH * OH);
    int oy  = rem / OH, ox = rem % OH;
    const _Float16* xb = x + (size_t)b * IC * IH * IH;
    v16h av = (kloc == 0) ? conv1_stage<0>(xb, oy, ox)
                          : conv1_stage<16>(xb, oy, ox);
    *(v16h*)(lA + mloc * 32 + kloc) = av;
  }
  __syncthreads();

  const int mrow = lane & 15;
  const int half = lane >> 4;

  const _Float16* ab = lA + mrow * 32 + half * 8;
  v8h alo = *(const v8h*)(ab);
  v8h ahi = *(const v8h*)(ab + 16);
  v16h afrag = __builtin_shufflevector(alo, ahi,
      0, 1, 2, 3, 4, 5, 6, 7, 8, 9, 10, 11, 12, 13, 14, 15);

  v8f acc[4];
#pragma unroll
  for (int nt = 0; nt < 4; ++nt) {
    const _Float16* bb = w + (size_t)(nt * 16 + mrow) * 32 + half * 8;
    v16h bfrag = __builtin_shufflevector(*(const v8h*)(bb), *(const v8h*)(bb + 16),
        0, 1, 2, 3, 4, 5, 6, 7, 8, 9, 10, 11, 12, 13, 14, 15);
    v8f z = {};
    acc[nt] = __builtin_amdgcn_wmma_f32_16x16x32_f16(
        false, afrag, false, bfrag, (short)0, z, false, false);
  }

#pragma unroll
  for (int r = 0; r < 8; ++r) {
    int mm = m0 + r + half * 8;
    int bb = mm / (OH * OH);
    int rr = mm % (OH * OH);
    size_t pix = (((size_t)bb * OH + rr / OH) * OH + rr % OH) * OC;
#pragma unroll
    for (int nt = 0; nt < 4; ++nt) {
      float v = acc[nt][r];
      y[pix + nt * 16 + mrow] = f32_to_e4m3(v > 0.f ? v : 0.f);
    }
  }
}

__global__ void k_conv2_wmma_fp8(const unsigned char* __restrict__ x,
                                 const unsigned char* __restrict__ w,
                                 _Float16* __restrict__ y) {
  constexpr int IH = 112, IC = 64, OH = 56, OC = 128, KP = 640;
  const int lane = threadIdx.x;
  const int n0   = (blockIdx.x & 3) * 32;
  const int m0   = (blockIdx.x >> 2) * 64;
  const int mrow = lane & 15;
  const int half = lane >> 4;

  const unsigned char* abase[4];
  int iy0[4], ix0[4];
#pragma unroll
  for (int mt = 0; mt < 4; ++mt) {
    int m = m0 + mt * 16 + mrow;
    int b = m / (OH * OH), rem = m % (OH * OH);
    int oy = rem / OH, ox = rem % OH;
    iy0[mt] = 2 * oy; ix0[mt] = 2 * ox;
    abase[mt] = x + (((size_t)b * IH + 2 * oy) * IH + 2 * ox) * IC + half * 16;
  }
  const unsigned char* wq0 = w + (size_t)(n0 + mrow) * KP + half * 16;
  const unsigned char* wq1 = wq0 + (size_t)16 * KP;

  auto loadA64 = [&](const unsigned char* q, v16i& d, int base) {
    int4 c0 = *(const int4*)(q);
    int4 c1 = *(const int4*)(q + 32);
    d[base + 0] = c0.x; d[base + 1] = c0.y;
    d[base + 2] = c0.z; d[base + 3] = c0.w;
    d[base + 4] = c1.x; d[base + 5] = c1.y;
    d[base + 6] = c1.z; d[base + 7] = c1.w;
  };
  auto loadB128 = [&](const unsigned char* q, v16i& d) {
    int4 c0 = *(const int4*)(q);
    int4 c1 = *(const int4*)(q + 32);
    int4 c2 = *(const int4*)(q + 64);
    int4 c3 = *(const int4*)(q + 96);
    d[0]  = c0.x; d[1]  = c0.y; d[2]  = c0.z; d[3]  = c0.w;
    d[4]  = c1.x; d[5]  = c1.y; d[6]  = c1.z; d[7]  = c1.w;
    d[8]  = c2.x; d[9]  = c2.y; d[10] = c2.z; d[11] = c2.w;
    d[12] = c3.x; d[13] = c3.y; d[14] = c3.z; d[15] = c3.w;
  };

  v8f acc[8] = {};
#pragma unroll
  for (int t2 = 0; t2 < 5; ++t2) {
    const int ta = 2 * t2, tb = 2 * t2 + 1;
    const int kya = ta / 3, kxa = ta % 3;
    const int kyb = tb / 3, kxb = tb % 3;
    const size_t offa = ((size_t)kya * IH + kxa) * IC;
    const size_t offb = ((size_t)kyb * IH + kxb) * IC;

    v16i a[4];
#pragma unroll
    for (int mt = 0; mt < 4; ++mt) {
      v16i av = {};
      if (iy0[mt] + kya < IH && ix0[mt] + kxa < IH)
        loadA64(abase[mt] + offa, av, 0);
      if (tb < 9 && iy0[mt] + kyb < IH && ix0[mt] + kxb < IH)
        loadA64(abase[mt] + offb, av, 8);
      a[mt] = av;
    }
    v16i b0 = {}, b1 = {};
    loadB128(wq0 + t2 * 128, b0);
    loadB128(wq1 + t2 * 128, b1);
    if (t2 < 4) __builtin_prefetch(wq0 + (t2 + 1) * 128, 0, 3);

#pragma unroll
    for (int mt = 0; mt < 4; ++mt) {
      acc[mt * 2 + 0] = __builtin_amdgcn_wmma_f32_16x16x128_fp8_fp8(
          a[mt], b0, (short)0, acc[mt * 2 + 0], false, false);
      acc[mt * 2 + 1] = __builtin_amdgcn_wmma_f32_16x16x128_fp8_fp8(
          a[mt], b1, (short)0, acc[mt * 2 + 1], false, false);
    }
  }

#pragma unroll
  for (int nt = 0; nt < 2; ++nt) {
    int oc = n0 + nt * 16 + mrow;
#pragma unroll
    for (int mt = 0; mt < 4; ++mt) {
#pragma unroll
      for (int r = 0; r < 8; ++r) {
        int mm = m0 + mt * 16 + r + half * 8;
        int bb = mm / (OH * OH);
        int rr = mm % (OH * OH);
        float v = acc[mt * 2 + nt][r];
        y[(((size_t)bb * OC + oc) * OH + rr / OH) * OH + rr % OH] =
            (_Float16)(v > 0.f ? v : 0.f);
      }
    }
  }
}

__global__ void k_head(const _Float16* __restrict__ x2, const float* __restrict__ fw,
                       const float* __restrict__ fb, float* __restrict__ out) {
  __shared__ float g[128];
  int b = blockIdx.x, c = threadIdx.x;
  const _Float16* p = x2 + ((size_t)b * 128 + c) * (56 * 56);
  float s = 0.f;
  for (int i = 0; i < 56 * 56; ++i) s += (float)p[i];
  g[c] = s * (1.f / (56.f * 56.f));
  __syncthreads();
  if (c < NCLSC) {
    float a = fb[c];
    for (int cc = 0; cc < 128; ++cc) a += g[cc] * fw[cc * NCLSC + c];
    out[b * NCLSC + c] = a;
  }
}

extern "C" void kernel_launch(void* const* d_in, const int* in_sizes, int n_in,
                              void* d_out, int out_size, void* d_ws, size_t ws_size,
                              hipStream_t stream) {
  (void)in_sizes; (void)n_in; (void)out_size; (void)ws_size;
  const float* F64  = (const float*)d_in[0];
  const float* F224 = (const float*)d_in[1];
  const float* sw1  = (const float*)d_in[2];
  const float* sw2  = (const float*)d_in[3];
  const float* pw   = (const float*)d_in[4];
  const float* pb   = (const float*)d_in[5];
  const float* ew1  = (const float*)d_in[6];
  const float* ew2  = (const float*)d_in[7];
  const float* fw   = (const float*)d_in[8];
  const float* fb   = (const float*)d_in[9];
  float* out = (float*)d_out;

  char* ws = (char*)d_ws;
  size_t off = 0;
  auto alloc = [&](size_t bytes) -> void* {
    void* p = ws + off;
    off = (off + bytes + 255) & ~(size_t)255;
    return p;
  };
  float*         s_in = (float*)        alloc((size_t)BB * 3 * 64 * 64 * 4);
  float*         h1   = (float*)        alloc((size_t)BB * 32 * 32 * 32 * 4);
  float*         h2   = (float*)        alloc((size_t)BB * 64 * 16 * 16 * 4);
  int*           sidx = (int*)          alloc((size_t)BB * NSELC * 4);
  _Float16*      w1h  = (_Float16*)     alloc((size_t)64 * 32 * 2);
  unsigned char* w2q  = (unsigned char*)alloc((size_t)128 * 640);
  _Float16*      evin = (_Float16*)     alloc((size_t)BB * 3 * 224 * 224 * 2);
  unsigned char* x1q  = (unsigned char*)alloc((size_t)BB * 112 * 112 * 64);
  _Float16*      x2h  = (_Float16*)     alloc((size_t)BB * 128 * 56 * 56 * 2);

  k_cvt_weights<<<(64 * 32 + 128 * 640 + 255) / 256, 256, 0, stream>>>(ew1, ew2, w1h, w2q);
  k_skim_resize<<<(BB * 3 * 64 * 64 + 255) / 256, 256, 0, stream>>>(F64, s_in);
  k_sconv1<<<(BB * 32 * 32 * 32 + 255) / 256, 256, 0, stream>>>(s_in, sw1, h1);
  k_sconv2<<<(BB * 64 * 16 * 16 + 255) / 256, 256, 0, stream>>>(h1, sw2, h2);
  k_policy<<<BB, 64, 0, stream>>>(h2, pw, pb, sidx);
  k_eval_resize<<<(BB * 3 * 224 * 224 + 255) / 256, 256, 0, stream>>>(F224, sidx, evin);

  k_conv1_wmma<<<12544, 32, 0, stream>>>(evin, w1h, x1q);
  k_conv2_wmma_fp8<<<784 * 4, 32, 0, stream>>>(x1q, w2q, x2h);

  k_head<<<BB, 128, 0, stream>>>(x2h, fw, fb, out);
}
